// GRUAttentionPooling_89232240542331
// MI455X (gfx1250) — matrix operand order repaired and checked by reading, not compiled, not run on hardware
//
#include <hip/hip_runtime.h>
#include <math.h>

typedef __attribute__((ext_vector_type(16))) __bf16 v16bf;
typedef __attribute__((ext_vector_type(8)))  __bf16 v8bf;
typedef __attribute__((ext_vector_type(8)))  float  v8f;
typedef __attribute__((ext_vector_type(4)))  float  v4f;

#define H        128
#define NPG      512          // nodes per graph (serial scan length)
#define GPB      16           // graphs per block (WMMA M dimension)
#define NBLK     16           // 256 graphs / 16
#define NTILES   24           // 384 / 16 gate-column tiles
#define THREADS  256          // 8 waves * 32

// ---- LDS layout (bytes) ----
#define OFF_WIH   0                      // W_ih^T bf16 B-fragments: 24*4*32*16*2 = 98304
#define OFF_WHH   98304                  // W_hh^T bf16 B-fragments: 98304
#define OFF_XF    196608                 // x  bf16 A-fragments: 4*32*16*2 = 4096
#define OFF_HF    200704                 // h  bf16 A-fragments: 4096
#define OFF_RZ    204800                 // r/z pre-activations [16][256] f32 = 16384
#define OFF_INN   221184                 // gi part of n-gate [16][128] = 8192
#define OFF_HN    229376                 // gh part of n-gate [16][128] = 8192
#define SMEM_BYTES 237568

__global__ __launch_bounds__(THREADS, 1)
void gru_attn_pool_kernel(const float* __restrict__ x,
                          const float* __restrict__ W_ih,
                          const float* __restrict__ W_hh,
                          const float* __restrict__ b_ih,
                          const float* __restrict__ b_hh,
                          const float* __restrict__ W_att,
                          const float* __restrict__ b_att,
                          float* __restrict__ out)
{
  extern __shared__ char smem[];
  __bf16* wih = (__bf16*)(smem + OFF_WIH);
  __bf16* whh = (__bf16*)(smem + OFF_WHH);
  __bf16* xf  = (__bf16*)(smem + OFF_XF);
  __bf16* hf  = (__bf16*)(smem + OFF_HF);
  float*  rz  = (float*)(smem + OFF_RZ);
  float*  inn = (float*)(smem + OFF_INN);
  float*  hng = (float*)(smem + OFF_HN);

  const int tid  = threadIdx.x;
  const int g0   = blockIdx.x * GPB;
  const int lane = tid & 31;
  const int wv   = tid >> 5;          // wave id: owns tiles wv, wv+8, wv+16
  const int ma   = lane & 15;         // A-frag row / C-frag column
  const int hl   = lane >> 4;
  const int gm   = tid >> 4;          // gate-phase graph row (0..15)
  const int gl   = tid & 15;          // gate-phase sub-block (8 contiguous cols)
  const int ib   = gl * 8;            // first hidden index owned by this thread

  // ---- one-time: pack W_ih^T / W_hh^T into bf16 WMMA B-fragment layout ----
  // B (32x16, KxN): lane = N%16 (+16 for K-half 1), 16 packed bf16 = 16 consecutive K.
  for (int idx = tid; idx < NTILES*4*32*16; idx += THREADS) {
    int j    = idx & 15;
    int ln   = (idx >> 4) & 31;
    int c    = (idx >> 9) & 3;          // K chunk of 32
    int tile = idx >> 11;               // N tile of 16
    int n    = tile*16 + (ln & 15);     // gate column (row of W)
    int k    = c*32 + (ln >> 4)*16 + j; // input dim (col of W)
    wih[idx] = (__bf16)W_ih[n*H + k];
    whh[idx] = (__bf16)W_hh[n*H + k];
  }

  // biases, folded into the post-WMMA store adds
  // tiles wv, wv+8 -> r/z columns (combined): need b_ih+b_hh
  // tile  wv+16    -> n-gate columns: need them separately
  float bsum0 = b_ih[wv*16 + ma]        + b_hh[wv*16 + ma];
  float bsum1 = b_ih[(wv+8)*16 + ma]    + b_hh[(wv+8)*16 + ma];
  float bihn  = b_ih[(wv+16)*16 + ma];
  float bhhn  = b_hh[(wv+16)*16 + ma];

  // ---- register-resident per-thread state (8 hidden elems each) ----
  float hreg[8], acc[8], wt[8];
  #pragma unroll
  for (int q = 0; q < 8; ++q) { hreg[q] = 0.f; acc[q] = 0.f; wt[q] = W_att[ib + q]; }
  float mr = -__builtin_inff(), sr = 0.f;
  const float batt = b_att[0];

  // A-fragment slot for this thread's 8 contiguous K values (same for x and h):
  //   c = ib>>5, K-half = (gl>>1)&1, j-base = (gl&1)*8, frag lane = gm (+16 for K-half 1)
  const int laddr = (((gl >> 2)*32) + gm + ((gl >> 1) & 1)*16)*16 + (gl & 1)*8;

  // clear h fragments (h0 = 0) and seed x fragments for step 0
  for (int i = tid; i < 4*32*16; i += THREADS) hf[i] = (__bf16)0.f;
  {
    const float* xp = x + ((size_t)(g0 + gm)*NPG + 0)*H + ib;
    v4f a0 = *(const v4f*)xp;
    v4f a1 = *(const v4f*)(xp + 4);
    v8bf xb;
    #pragma unroll
    for (int q = 0; q < 4; ++q) { xb[q] = (__bf16)a0[q]; xb[4+q] = (__bf16)a1[q]; }
    *(v8bf*)(xf + laddr) = xb;
  }
  __syncthreads();

  for (int step = 0; step < NPG; ++step) {
    // -- prefetch next step's x row chunk into registers (off critical path) --
    int sn = (step + 1 < NPG) ? step + 1 : NPG - 1;
    const float* xp = x + ((size_t)(g0 + gm)*NPG + sn)*H + ib;
    v4f xr0 = *(const v4f*)xp;
    v4f xr1 = *(const v4f*)(xp + 4);

    // -- Phase B: WMMA  gi = x@Wih^T,  gh = h@Whh^T (C starts at inline 0) --
    v16bf xfr[4], hfr[4];
    #pragma unroll
    for (int c = 0; c < 4; ++c) {
      xfr[c] = *(const v16bf*)(xf + (c*32 + lane)*16);
      hfr[c] = *(const v16bf*)(hf + (c*32 + lane)*16);
    }
    #pragma unroll
    for (int t = 0; t < 3; ++t) {
      const int tile = wv + t*8;                // t=0,1: r/z tiles; t=2: n tile
      v8f ai = {}; v8f ah = {};
      #pragma unroll
      for (int c = 0; c < 4; ++c) {
        v16bf wi = *(const v16bf*)(wih + (size_t)((tile*4 + c)*32 + lane)*16);
        v16bf wh = *(const v16bf*)(whh + (size_t)((tile*4 + c)*32 + lane)*16);
        ai = __builtin_amdgcn_wmma_f32_16x16x32_bf16(false, xfr[c], false, wi,
                                                     (short)0, ai, false, false);
        ah = __builtin_amdgcn_wmma_f32_16x16x32_bf16(false, hfr[c], false, wh,
                                                     (short)0, ah, false, false);
      }
      // C/D layout: lane -> column, VGPR j -> row (+8 for lanes 16..31)
      if (t < 2) {                              // compile-time uniform: r/z sum
        const int   col = tile*16 + ma;
        const float bs  = (t == 0) ? bsum0 : bsum1;
        #pragma unroll
        for (int j = 0; j < 8; ++j)
          rz[(j + 8*hl)*(2*H) + col] = ai[j] + ah[j] + bs;
      } else {                                  // n-gate: keep gi/gh separate
        const int cc = tile*16 + ma - 2*H;
        #pragma unroll
        for (int j = 0; j < 8; ++j) {
          inn[(j + 8*hl)*H + cc] = ai[j] + bihn;
          hng[(j + 8*hl)*H + cc] = ah[j] + bhhn;
        }
      }
    }
    __syncthreads();

    // -- Phase C: GRU gates + online-softmax pooling (all state in registers) --
    v4f rv0 = *(const v4f*)(rz + gm*(2*H) + ib);
    v4f rv1 = *(const v4f*)(rz + gm*(2*H) + ib + 4);
    v4f zv0 = *(const v4f*)(rz + gm*(2*H) + H + ib);
    v4f zv1 = *(const v4f*)(rz + gm*(2*H) + H + ib + 4);
    v4f iv0 = *(const v4f*)(inn + gm*H + ib);
    v4f iv1 = *(const v4f*)(inn + gm*H + ib + 4);
    v4f gv0 = *(const v4f*)(hng + gm*H + ib);
    v4f gv1 = *(const v4f*)(hng + gm*H + ib + 4);

    float hnew[8];
    float lp = 0.f;
    #pragma unroll
    for (int q = 0; q < 8; ++q) {
      float rp = (q < 4) ? rv0[q & 3] : rv1[q & 3];
      float zp = (q < 4) ? zv0[q & 3] : zv1[q & 3];
      float ip = (q < 4) ? iv0[q & 3] : iv1[q & 3];
      float gp = (q < 4) ? gv0[q & 3] : gv1[q & 3];
      float r  = 1.f / (1.f + __expf(-rp));
      float z  = 1.f / (1.f + __expf(-zp));
      float nn = tanhf(ip + r*gp);
      float hv = (1.f - z)*nn + z*hreg[q];
      hnew[q] = hv;
      hreg[q] = hv;
      lp += hv * wt[q];
    }
    #pragma unroll
    for (int off = 8; off >= 1; off >>= 1)   // butterfly: sum lands in ALL 16 lanes
      lp += __shfl_xor(lp, off, 32);
    {
      float l  = lp + batt;
      float mn = fmaxf(mr, l);
      float sc = __expf(mr - mn);
      float pp = __expf(l - mn);
      sr = sr*sc + pp;
      mr = mn;
      #pragma unroll
      for (int q = 0; q < 8; ++q) acc[q] = acc[q]*sc + pp*hnew[q];
    }

    // -- write h_new and next x as bf16 A-fragments (one b128 store each) --
    {
      v8bf hb, xb;
      #pragma unroll
      for (int q = 0; q < 8; ++q) hb[q] = (__bf16)hnew[q];
      #pragma unroll
      for (int q = 0; q < 4; ++q) { xb[q] = (__bf16)xr0[q]; xb[4+q] = (__bf16)xr1[q]; }
      *(v8bf*)(hf + laddr) = hb;
      *(v8bf*)(xf + laddr) = xb;
    }
    __syncthreads();
  }

  // ---- epilogue: out = acc / sum (pure register state) ----
  #pragma unroll
  for (int q = 0; q < 8; ++q)
    out[(size_t)(g0 + gm)*H + ib + q] = acc[q] / sr;
}

extern "C" void kernel_launch(void* const* d_in, const int* in_sizes, int n_in,
                              void* d_out, int out_size, void* d_ws, size_t ws_size,
                              hipStream_t stream) {
  const float* x     = (const float*)d_in[0];
  // d_in[1] = batch (int64): sorted equal-size segments guaranteed -> unused
  const float* W_ih  = (const float*)d_in[2];
  const float* W_hh  = (const float*)d_in[3];
  const float* b_ih  = (const float*)d_in[4];
  const float* b_hh  = (const float*)d_in[5];
  const float* W_att = (const float*)d_in[6];
  const float* b_att = (const float*)d_in[7];
  float* out = (float*)d_out;

  gru_attn_pool_kernel<<<dim3(NBLK), dim3(THREADS), SMEM_BYTES, stream>>>(
      x, W_ih, W_hh, b_ih, b_hh, W_att, b_att, out);
}
